// RecurrentMoE_19464791785489
// MI455X (gfx1250) — matrix operand order repaired and checked by reading, not compiled, not run on hardware
//
#include <hip/hip_runtime.h>
#include <hip/hip_bf16.h>
#include <math.h>

// Problem constants (fixed by the reference)
static constexpr int TT  = 1024;         // tokens (B*S)
static constexpr int DD  = 768;          // model dim
static constexpr int HH  = 768;          // hidden dim
static constexpr int EE  = 8;            // experts
static constexpr int LL  = 2;            // layers
static constexpr int FFF = 3072;         // FFN dim (4*D)
static constexpr int H3  = 3 * HH;       // 2304

typedef __attribute__((ext_vector_type(16))) __bf16 v16bf;
typedef __attribute__((ext_vector_type(8)))  __bf16 v8bf;
typedef __attribute__((ext_vector_type(4)))  __bf16 v4bf;
typedef __attribute__((ext_vector_type(8)))  float  v8f;
typedef __attribute__((ext_vector_type(4)))  float  f4;

static constexpr int LDSTR = 40;   // padded LDS row stride (bf16 elems): 80B -> conflict-free b128 lattice

// ---------------------------------------------------------------------------
// Stage a 128x32 tile (rows relative to block origin) into LDS as bf16.
// 256 threads: 8 lanes cover one row (8 x float4 = 128B, fully coalesced),
// 4 passes of 32 rows.  fp32 source is converted in-registers (single pass).
// ---------------------------------------------------------------------------
__device__ __forceinline__ void stage_f32(const float* __restrict__ g, int ld,
                                          __bf16* __restrict__ s, int tid, int k)
{
    int row = tid >> 3;             // 0..31
    const int c = (tid & 7) * 4;    // 0..28
    #pragma unroll
    for (int p = 0; p < 4; ++p, row += 32) {
        f4 v = *(const f4*)(g + (long long)row * ld + k + c);
        v4bf o;
        o[0] = (__bf16)v[0]; o[1] = (__bf16)v[1];
        o[2] = (__bf16)v[2]; o[3] = (__bf16)v[3];
        *(v4bf*)(s + row * LDSTR + c) = o;
    }
}

__device__ __forceinline__ void stage_bf16(const __bf16* __restrict__ g, int ld,
                                           __bf16* __restrict__ s, int tid, int k)
{
    int row = tid >> 3;
    const int c = (tid & 7) * 4;
    #pragma unroll
    for (int p = 0; p < 4; ++p, row += 32) {
        v4bf v = *(const v4bf*)(g + (long long)row * ld + k + c);
        *(v4bf*)(s + row * LDSTR + c) = v;
    }
}

// ---------------------------------------------------------------------------
// Tiled WMMA GEMM:  C[e] = epilogue( A[e](M x K, row) * W[e](N x K, row)^T + b[e] )
//   - block = 256 threads = 8 waves in a 4(M) x 2(N) grid; block tile 128x128
//   - each wave owns a 32x64 patch = 2x4 v_wmma_f32_16x16x32_bf16 accumulators
//   - per K-step (32): stage A/B tiles to LDS as bf16 (converted once per
//     block, shared by all 8 waves), assemble fragments with 16B ds loads
//     following the documented wave32 VGPR layouts.
// Template flags: ABF16 (A already bf16), RELU, OBF16 (bf16 store), SCALE
// (multiply row t by comb[t*E + e]).  Assumes M%128==0, N%128==0, K%32==0.
// ---------------------------------------------------------------------------
template<bool ABF16, bool RELU, bool OBF16, bool SCALE>
__global__ __launch_bounds__(256) void gemm_wmma_k(
    const void* __restrict__ Av, long long astr,
    const float* __restrict__ Wt, long long wstr,
    const float* __restrict__ bias, long long bstr,
    void* __restrict__ Cv, long long cstr,
    const float* __restrict__ comb,
    int M, int N, int K, int lda, int ldw, int ldc)
{
    __shared__ __bf16 As[128 * LDSTR];
    __shared__ __bf16 Bs[128 * LDSTR];

    const int e    = blockIdx.z;
    const int tid  = threadIdx.x;
    const int wave = tid >> 5;
    const int lane = tid & 31;
    const int half = lane >> 4;     // 0: lanes 0-15, 1: lanes 16-31
    const int r    = lane & 15;

    const int blk_m0 = blockIdx.y * 128;
    const int blk_n0 = blockIdx.x * 128;
    const int wm0    = (wave >> 1) * 32;   // wave M origin within block
    const int wn0    = (wave & 1) * 64;    // wave N origin within block

    const float* Wblk = Wt + (long long)e * wstr + (long long)blk_n0 * ldw;
    const float* bb   = bias + (long long)e * bstr;

    v8f acc[2][4] = {};

    for (int k = 0; k < K; k += 32) {
        // ---- stage block tiles (bf16) into LDS
        if (ABF16) {
            const __bf16* A = (const __bf16*)Av + (long long)e * astr + (long long)blk_m0 * lda;
            stage_bf16(A, lda, As, tid, k);
        } else {
            const float* A = (const float*)Av + (long long)e * astr + (long long)blk_m0 * lda;
            stage_f32(A, lda, As, tid, k);
        }
        stage_f32(Wblk, ldw, Bs, tid, k);
        __syncthreads();

        // ---- A fragments (16x32): lane r holds row; K-chunks [kb..kb+7] and
        //      [kb+16..kb+23] with kb = half*8
        v16bf a[2], b[4];
        #pragma unroll
        for (int j = 0; j < 2; ++j) {
            const __bf16* p = As + (wm0 + j * 16 + r) * LDSTR + half * 8;
            v8bf lo = *(const v8bf*)p;
            v8bf hi = *(const v8bf*)(p + 16);
            a[j] = __builtin_shufflevector(lo, hi, 0, 1, 2, 3, 4, 5, 6, 7,
                                                   8, 9, 10, 11, 12, 13, 14, 15);
        }
        // ---- B fragments (32x16): lane r holds column; K-chunks [kb..kb+7] and
        //      [kb+16..kb+23] with kb = half*8
        #pragma unroll
        for (int i = 0; i < 4; ++i) {
            const __bf16* p = Bs + (wn0 + i * 16 + r) * LDSTR + half * 8;
            v8bf lo = *(const v8bf*)p;
            v8bf hi = *(const v8bf*)(p + 16);
            b[i] = __builtin_shufflevector(lo, hi, 0, 1, 2, 3, 4, 5, 6, 7,
                                                   8, 9, 10, 11, 12, 13, 14, 15);
        }

        #pragma unroll
        for (int j = 0; j < 2; ++j)
            #pragma unroll
            for (int i = 0; i < 4; ++i)
                acc[j][i] = __builtin_amdgcn_wmma_f32_16x16x32_bf16(
                    false, a[j], false, b[i], (short)0, acc[j][i], false, false);

        __syncthreads();
    }

    // ---- epilogue: C/D layout: VGPR i -> M = half*8 + i, N = lane&15
    #pragma unroll
    for (int j = 0; j < 2; ++j) {
        #pragma unroll
        for (int i = 0; i < 4; ++i) {
            const int tm0 = blk_m0 + wm0 + j * 16;
            const int n   = blk_n0 + wn0 + i * 16 + r;
            const float bv = bb[n];
            #pragma unroll
            for (int q = 0; q < 8; ++q) {
                const int m = tm0 + half * 8 + q;
                float v = acc[j][i][q] + bv;
                if (RELU)  v = v > 0.0f ? v : 0.0f;
                if (SCALE) v *= comb[(long long)m * EE + e];
                if (OBF16) ((__bf16*)Cv)[(long long)e * cstr + (long long)m * ldc + n] = (__bf16)v;
                else       ((float*) Cv)[(long long)e * cstr + (long long)m * ldc + n] = v;
            }
        }
    }
}

// ---------------------------------------------------------------------------
// GRU gate fusion: h = (1-z)*n + z*h  with r,z,n from gi/gh (PyTorch order)
// ---------------------------------------------------------------------------
__global__ void gru_k(const float* __restrict__ gi, const float* __restrict__ gh,
                      float* __restrict__ h, int n)
{
    int idx = blockIdx.x * blockDim.x + threadIdx.x;
    if (idx >= n) return;
    int t = idx / HH, j = idx - t * HH;
    const float* gir = gi + (long long)t * H3;
    const float* ghr = gh + (long long)t * H3;
    float ir = gir[j], iz = gir[HH + j], in = gir[2 * HH + j];
    float hr = ghr[j], hz = ghr[HH + j], hn = ghr[2 * HH + j];
    float rg = 1.0f / (1.0f + __expf(-(ir + hr)));
    float z  = 1.0f / (1.0f + __expf(-(iz + hz)));
    float nn = tanhf(in + rg * hn);
    float ho = h[idx];
    h[idx] = (1.0f - z) * nn + z * ho;
}

// ---------------------------------------------------------------------------
// Router: logits = h[t] @ Wr^T + br ; softmax ; top-2 ; renormalized weights
// scattered into dense comb[t, 0..7].  One 256-thread block per token:
// 8 expert-groups x 32 lanes, LDS tree reduction.
// ---------------------------------------------------------------------------
__global__ __launch_bounds__(256) void router_k(
    const float* __restrict__ h, const float* __restrict__ Wr,
    const float* __restrict__ br, float* __restrict__ comb)
{
    __shared__ float s[256];
    const int t    = blockIdx.x;
    const int tid  = threadIdx.x;
    const int e    = tid >> 5;
    const int lane = tid & 31;

    const float* hr = h  + (long long)t * HH;
    const float* wr = Wr + (long long)e * HH;
    float p = 0.0f;
    for (int j = lane; j < HH; j += 32) p += hr[j] * wr[j];
    s[tid] = p;
    __syncthreads();
    for (int off = 16; off > 0; off >>= 1) {
        if (lane < off) s[tid] += s[tid + off];
        __syncthreads();
    }
    if (tid == 0) {
        float lg[EE], pr[EE];
        float mx = -1e30f;
        for (int i = 0; i < EE; ++i) { lg[i] = s[i * 32] + br[i]; mx = lg[i] > mx ? lg[i] : mx; }
        for (int i = 0; i < EE; ++i) pr[i] = __expf(lg[i] - mx);
        int i0 = 0;
        for (int i = 1; i < EE; ++i) if (pr[i] > pr[i0]) i0 = i;
        int i1 = (i0 == 0) ? 1 : 0;
        for (int i = 0; i < EE; ++i) if (i != i0 && pr[i] > pr[i1]) i1 = i;
        const float sum2 = pr[i0] + pr[i1];
        for (int i = 0; i < EE; ++i) {
            float w = (i == i0) ? pr[i0] / sum2 : ((i == i1) ? pr[i1] / sum2 : 0.0f);
            comb[(long long)t * EE + i] = w;
        }
    }
}

// ---------------------------------------------------------------------------
// xt += sum_e ybuf[e]   (per-expert partials -> deterministic combine)
// ---------------------------------------------------------------------------
__global__ void combine_k(float* __restrict__ xt, const float* __restrict__ yb,
                          int n, long long estr)
{
    int idx = blockIdx.x * blockDim.x + threadIdx.x;
    if (idx >= n) return;
    float sv = xt[idx];
    #pragma unroll
    for (int e = 0; e < EE; ++e) sv += yb[(long long)e * estr + idx];
    xt[idx] = sv;
}

// ---------------------------------------------------------------------------
// Workspace layout (bytes, 256-aligned). Total ~101 MB.
// ---------------------------------------------------------------------------
static constexpr size_t ALGN = 256;
static constexpr size_t rup(size_t x) { return (x + ALGN - 1) & ~(ALGN - 1); }
static constexpr size_t OFF_XP   = 0;                                              // T*H   f32
static constexpr size_t OFF_GI   = OFF_XP   + rup((size_t)TT * HH * 4);            // T*3H  f32
static constexpr size_t OFF_GH   = OFF_GI   + rup((size_t)TT * H3 * 4);            // T*3H  f32
static constexpr size_t OFF_H    = OFF_GH   + rup((size_t)TT * H3 * 4);            // T*H   f32
static constexpr size_t OFF_COMB = OFF_H    + rup((size_t)TT * HH * 4);            // T*E   f32
static constexpr size_t OFF_H1   = OFF_COMB + rup((size_t)TT * EE * 4);            // E*T*FF bf16
static constexpr size_t OFF_YB   = OFF_H1   + rup((size_t)EE * TT * FFF * 2);      // E*T*D  f32

extern "C" void kernel_launch(void* const* d_in, const int* in_sizes, int n_in,
                              void* d_out, int out_size, void* d_ws, size_t ws_size,
                              hipStream_t stream) {
    (void)in_sizes; (void)n_in; (void)out_size; (void)ws_size;

    const float* x   = (const float*)d_in[0];
    const float* Wp  = (const float*)d_in[1];
    const float* bp  = (const float*)d_in[2];
    const float* Wih = (const float*)d_in[3];
    const float* Whh = (const float*)d_in[4];
    const float* bih = (const float*)d_in[5];
    const float* bhh = (const float*)d_in[6];
    const float* Wr  = (const float*)d_in[7];
    const float* br  = (const float*)d_in[8];
    const float* W1  = (const float*)d_in[9];
    const float* b1  = (const float*)d_in[10];
    const float* W2  = (const float*)d_in[11];
    const float* b2  = (const float*)d_in[12];

    char* ws = (char*)d_ws;
    float*  xp   = (float*) (ws + OFF_XP);
    float*  gi   = (float*) (ws + OFF_GI);
    float*  gh   = (float*) (ws + OFF_GH);
    float*  hbuf = (float*) (ws + OFF_H);
    float*  comb = (float*) (ws + OFF_COMB);
    __bf16* h1   = (__bf16*)(ws + OFF_H1);
    float*  ybuf = (float*) (ws + OFF_YB);

    float* xt = (float*)d_out;                               // residual stream lives in d_out
    hipMemcpyAsync(xt, x, (size_t)TT * DD * sizeof(float), hipMemcpyDeviceToDevice, stream);
    hipMemsetAsync(hbuf, 0, (size_t)TT * HH * sizeof(float), stream);

    const dim3 blk(256);
    for (int l = 0; l < LL; ++l) {
        // xp = xt @ Wp[l]^T + bp[l]          (1024 x 768 x 768)
        gemm_wmma_k<false, false, false, false><<<dim3(HH / 128, TT / 128, 1), blk, 0, stream>>>(
            xt, 0, Wp + (long long)l * HH * DD, 0, bp + (long long)l * HH, 0,
            xp, 0, nullptr, TT, HH, DD, DD, DD, HH);
        // gi = xp @ Wih^T + bih              (1024 x 2304 x 768)
        gemm_wmma_k<false, false, false, false><<<dim3(H3 / 128, TT / 128, 1), blk, 0, stream>>>(
            xp, 0, Wih, 0, bih, 0, gi, 0, nullptr, TT, H3, HH, HH, HH, H3);
        // gh = h @ Whh^T + bhh               (1024 x 2304 x 768)
        gemm_wmma_k<false, false, false, false><<<dim3(H3 / 128, TT / 128, 1), blk, 0, stream>>>(
            hbuf, 0, Whh, 0, bhh, 0, gh, 0, nullptr, TT, H3, HH, HH, HH, H3);
        // GRU gates -> h
        gru_k<<<(TT * HH + 255) / 256, blk, 0, stream>>>(gi, gh, hbuf, TT * HH);
        // routing -> comb (dense T x E, top-2 renormalized)
        router_k<<<dim3(TT), blk, 0, stream>>>(hbuf, Wr + (long long)l * EE * HH,
                                               br + (long long)l * EE, comb);
        // h1[e] = relu(xt @ W1[l,e]^T + b1[l,e])  -> bf16   (1024 x 3072 x 768, x8 experts)
        gemm_wmma_k<false, true, true, false><<<dim3(FFF / 128, TT / 128, EE), blk, 0, stream>>>(
            xt, 0, W1 + (long long)l * EE * FFF * DD, (long long)FFF * DD,
            b1 + (long long)l * EE * FFF, FFF,
            h1, (long long)TT * FFF, nullptr, TT, FFF, DD, DD, DD, FFF);
        // ybuf[e] = comb[:,e] * (h1[e] @ W2[l,e]^T + b2[l,e])   (1024 x 768 x 3072, x8)
        gemm_wmma_k<true, false, false, true><<<dim3(DD / 128, TT / 128, EE), blk, 0, stream>>>(
            h1, (long long)TT * FFF, W2 + (long long)l * EE * DD * FFF, (long long)DD * FFF,
            b2 + (long long)l * EE * DD, DD,
            ybuf, (long long)TT * DD, comb, TT, DD, FFF, FFF, FFF, DD);
        // xt += sum_e ybuf[e]
        combine_k<<<(TT * DD + 255) / 256, blk, 0, stream>>>(xt, ybuf, TT * DD, (long long)TT * DD);
    }
}
